// MultiHeadAttention_15633680957666
// MI455X (gfx1250) — matrix operand order repaired and checked by reading, not compiled, not run on hardware
//
#include <hip/hip_runtime.h>
#include <hip/hip_bf16.h>

// MHA: B=2, L=4096, D=512, H=8, HD=64. ~86 GFLOP vs ~66MB traffic -> compute
// bound -> all GEMMs via v_wmma_f32_16x16x32_bf16 (wave32 CDNA5).
// All WMMA operands pre-staged in register-image layout so the hot loops are
// pure {b128 loads + wmma}; activations/weights converted to bf16 exactly once.
#define BB 2
#define LL 4096
#define DD 512
#define HH 8
#define HDD 64
#define OC 512          // H*HD
#define NROW (BB*LL)    // 8192

typedef __attribute__((ext_vector_type(16))) __bf16 v16bf;
typedef __attribute__((ext_vector_type(8)))  __bf16 bf16x8;
typedef __attribute__((ext_vector_type(8)))  float  v8f;

static __device__ __forceinline__ v8f wmma_bf16(v16bf a, v16bf b, v8f c) {
  return __builtin_amdgcn_wmma_f32_16x16x32_bf16(false, a, false, b, (short)0, c,
                                                 false, false);
}

// contiguous 16-bf16 chunk (32B): packed A/B register images
static __device__ __forceinline__ v16bf ldv16_c(const __bf16* p) {
  bf16x8 lo = *(const bf16x8*)p;
  bf16x8 hi = *(const bf16x8*)(p + 8);
  v16bf r;
#pragma unroll
  for (int i = 0; i < 8; ++i) { r[i] = lo[i]; r[8 + i] = hi[i]; }
  return r;
}
// A-operand from natural [row][k] storage: slots 0..7 at p[0..7], 8..15 at p[16..23]
static __device__ __forceinline__ v16bf ldv16_a(const __bf16* p) {
  bf16x8 lo = *(const bf16x8*)p;
  bf16x8 hi = *(const bf16x8*)(p + 16);
  v16bf r;
#pragma unroll
  for (int i = 0; i < 8; ++i) { r[i] = lo[i]; r[8 + i] = hi[i]; }
  return r;
}

static __device__ __forceinline__ float halfmax(float v) {
#pragma unroll
  for (int m = 1; m <= 8; m <<= 1) v = fmaxf(v, __shfl_xor(v, m, 32));
  return v;
}
static __device__ __forceinline__ float halfsum(float v) {
#pragma unroll
  for (int m = 1; m <= 8; m <<= 1) v += __shfl_xor(v, m, 32);
  return v;
}

// ---------------------------------------------------------------------------
// Kernel 0a: pack weights (fp32 KxOC) -> bf16 B-operand chunks.
// chunk ((tn*16+kstep)*32+lane): slot i <-> K = kstep*32+(lane>>4)*8+i (+8 for i>=8),
// col = tn*16+(lane&15).
// ---------------------------------------------------------------------------
__global__ __launch_bounds__(256)
void pack_w(const float* __restrict__ Wq, const float* __restrict__ Wk,
            const float* __restrict__ Wv, const float* __restrict__ Wo,
            __bf16* __restrict__ pq, __bf16* __restrict__ pk,
            __bf16* __restrict__ pv, __bf16* __restrict__ po) {
  const float* W; __bf16* P;
  if (blockIdx.y == 0)      { W = Wq; P = pq; }
  else if (blockIdx.y == 1) { W = Wk; P = pk; }
  else if (blockIdx.y == 2) { W = Wv; P = pv; }
  else                      { W = Wo; P = po; }

  const int t     = blockIdx.x * 256 + threadIdx.x;  // 0..16383
  const int lane  = t & 31;
  const int kstep = (t >> 5) & 15;
  const int tn    = t >> 9;
  const int col   = tn * 16 + (lane & 15);
  const int kbase = kstep * 32 + (lane >> 4) * 8;

  v16bf v;
#pragma unroll
  for (int i = 0; i < 16; ++i) v[i] = (__bf16)W[(size_t)(kbase + (i < 8 ? i : i + 8)) * OC + col];
  *(v16bf*)(P + (size_t)t * 16) = v;
}

// ---------------------------------------------------------------------------
// Kernel 0b: pack activations (fp32 [8192,512]) -> bf16 A-operand chunks.
// chunk ((tm*16+kstep)*32+lane): row = tm*16+(lane&15);
// slots 0..7 <-> k = kstep*32+half*8+i, slots 8..15 <-> +16. Paid once;
// removes all v_cvt from the projection inner loop (was converted 32x).
// ---------------------------------------------------------------------------
__global__ __launch_bounds__(256)
void pack_x(const float* __restrict__ Xq, const float* __restrict__ Xk,
            const float* __restrict__ Xv,
            __bf16* __restrict__ xq, __bf16* __restrict__ xk,
            __bf16* __restrict__ xv) {
  const float* X; __bf16* P;
  if (blockIdx.y == 0)      { X = Xq; P = xq; }
  else if (blockIdx.y == 1) { X = Xk; P = xk; }
  else                      { X = Xv; P = xv; }

  const int t     = blockIdx.x * 256 + threadIdx.x;  // 0..262143
  const int lane  = t & 31;
  const int kstep = (t >> 5) & 15;
  const int tm    = t >> 9;                          // 0..511
  const int row   = lane & 15;
  const int half  = lane >> 4;
  const int k0    = kstep * 32;
  const float* src = X + (size_t)(tm * 16 + row) * DD + k0 + half * 8;

  const float4 f0 = *(const float4*)(src);
  const float4 f1 = *(const float4*)(src + 4);
  const float4 f2 = *(const float4*)(src + 16);
  const float4 f3 = *(const float4*)(src + 20);
  v16bf v;
  v[0] = (__bf16)f0.x;  v[1] = (__bf16)f0.y;  v[2] = (__bf16)f0.z;  v[3] = (__bf16)f0.w;
  v[4] = (__bf16)f1.x;  v[5] = (__bf16)f1.y;  v[6] = (__bf16)f1.z;  v[7] = (__bf16)f1.w;
  v[8] = (__bf16)f2.x;  v[9] = (__bf16)f2.y;  v[10] = (__bf16)f2.z; v[11] = (__bf16)f2.w;
  v[12] = (__bf16)f3.x; v[13] = (__bf16)f3.y; v[14] = (__bf16)f3.z; v[15] = (__bf16)f3.w;
  *(v16bf*)(P + (size_t)t * 16) = v;
}

// ---------------------------------------------------------------------------
// Kernel 1: QKV projection. One wave -> one 16x16 tile, K=512 (16 WMMAs).
// Inner loop: 2x b128 (packed A) + 2x b128 (packed B) + 1 wmma. No VALU.
// Q,K written [b,h,l,hd]; V written TRANSPOSED [b,h,hd,l].
// ---------------------------------------------------------------------------
__global__ __launch_bounds__(256)
void mha_proj_qkv(const __bf16* __restrict__ xq, const __bf16* __restrict__ xk,
                  const __bf16* __restrict__ xv,
                  const __bf16* __restrict__ pq, const float* __restrict__ bq,
                  const __bf16* __restrict__ pk, const float* __restrict__ bk,
                  const __bf16* __restrict__ pv, const float* __restrict__ bv,
                  __bf16* __restrict__ qd, __bf16* __restrict__ kd,
                  __bf16* __restrict__ vd) {
  const int lane = threadIdx.x & 31;
  const int wv   = threadIdx.x >> 5;
  const int tile = blockIdx.x * 8 + wv;   // 512*32 tiles
  const int tn = tile & 31;
  const int tm = tile >> 5;

  const __bf16* Xp; const __bf16* P; const float* bias; __bf16* dst;
  bool transpose_out = false;
  if (blockIdx.y == 0)      { Xp = xq; P = pq; bias = bq; dst = qd; }
  else if (blockIdx.y == 1) { Xp = xk; P = pk; bias = bk; dst = kd; }
  else                      { Xp = xv; P = pv; bias = bv; dst = vd; transpose_out = true; }

  const int row  = lane & 15;
  const int half = lane >> 4;
  const int col  = tn * 16 + row;
  const __bf16* Ab = Xp + ((size_t)tm * 16 * 32 + lane) * 16;
  const __bf16* Bb = P  + ((size_t)tn * 16 * 32 + lane) * 16;

  v8f acc = {};
#pragma unroll 4
  for (int s = 0; s < 16; ++s) {
    const v16bf a = ldv16_c(Ab + (size_t)s * 32 * 16);
    const v16bf b = ldv16_c(Bb + (size_t)s * 32 * 16);
    acc = wmma_bf16(a, b, acc);
  }

  const float bia = bias[col];
  const int hh = col >> 6;
  const int hd = col & 63;
#pragma unroll
  for (int r = 0; r < 8; ++r) {          // C layout: vgpr r -> row r + 8*half
    const int n  = tm * 16 + r + 8 * half;
    const int bb = n >> 12;
    const int ll = n & (LL - 1);
    const __bf16 val = (__bf16)(acc[r] + bia);
    if (transpose_out)
      dst[((size_t)((bb * HH + hh) * HDD + hd) * LL) + ll] = val;     // [b,h,hd,l]
    else
      dst[(((size_t)(bb * HH + hh) * LL + ll) << 6) + hd] = val;      // [b,h,l,hd]
  }
}

// ---------------------------------------------------------------------------
// Kernel 2: flash attention. One wave per 16-row Q tile; 64 keys per iter
// (16 WMMAs / iter) so softmax shuffles + barriers amortize over 2x the keys.
// ---------------------------------------------------------------------------
__global__ __launch_bounds__(256)
void mha_flash(const __bf16* __restrict__ qbf, const __bf16* __restrict__ kbf,
               const __bf16* __restrict__ vT, const int* __restrict__ mask,
               __bf16* __restrict__ abf) {
  __shared__ __bf16 pbuf[8][16 * 64];   // per-wave P buffer (16 KB)

  const int lane = threadIdx.x & 31;
  const int wv   = threadIdx.x >> 5;
  const int tile = blockIdx.x * 8 + wv;   // 0..4095
  const int qt   = tile & 255;
  const int bh   = tile >> 8;
  const int bat  = bh >> 3;
  const int hed  = bh & 7;
  const int row  = lane & 15;
  const int half = lane >> 4;

  const __bf16* Qp = qbf + ((size_t)bh * LL + qt * 16) * HDD;
  const __bf16* Kp = kbf + (size_t)bh * LL * HDD;
  const __bf16* Vt = vT  + (size_t)bh * HDD * LL;   // [hd][l]

  v16bf aq[2];
#pragma unroll
  for (int part = 0; part < 2; ++part)
    aq[part] = ldv16_c(Qp + row * HDD + part * 32 + half * 16);

  v8f o0 = {}, o1 = {}, o2 = {}, o3 = {};
  float mrow[8], lrow[8];
#pragma unroll
  for (int r = 0; r < 8; ++r) { mrow[r] = -1e30f; lrow[r] = 0.0f; }

  for (int kt = 0; kt < LL / 64; ++kt) {
    if (kt + 1 < LL / 64) {
      __builtin_prefetch((const void*)(Kp + (size_t)(kt + 1) * 64 * HDD), 0, 3);
      __builtin_prefetch((const void*)(Kp + (size_t)(kt + 1) * 64 * HDD + 2048), 0, 3);
    }

    // ---- S = Q K^T : four 16x16 C tiles (64 keys)
    v8f s0 = {}, s1 = {}, s2 = {}, s3 = {};
#pragma unroll
    for (int t = 0; t < 4; ++t) {
      const int key = kt * 64 + t * 16 + row;      // B column for this lane
#pragma unroll
      for (int part = 0; part < 2; ++part) {
        const v16bf bk = ldv16_c(Kp + (size_t)key * HDD + part * 32 + half * 16);
        if (t == 0)      s0 = wmma_bf16(aq[part], bk, s0);
        else if (t == 1) s1 = wmma_bf16(aq[part], bk, s1);
        else if (t == 2) s2 = wmma_bf16(aq[part], bk, s2);
        else             s3 = wmma_bf16(aq[part], bk, s3);
      }
    }

    // ---- scale + mask
    const int key0 = kt * 64 + row;
    const bool ok0 = mask[bat * LL + key0] != 0;
    const bool ok1 = mask[bat * LL + key0 + 16] != 0;
    const bool ok2 = mask[bat * LL + key0 + 32] != 0;
    const bool ok3 = mask[bat * LL + key0 + 48] != 0;
    const float NEG = -__builtin_inff();
#pragma unroll
    for (int r = 0; r < 8; ++r) {
      s0[r] = ok0 ? s0[r] * 0.125f : NEG;          // 1/sqrt(64)
      s1[r] = ok1 ? s1[r] * 0.125f : NEG;
      s2[r] = ok2 ? s2[r] * 0.125f : NEG;
      s3[r] = ok3 ? s3[r] * 0.125f : NEG;
    }

    // ---- online softmax (row r+8*half resident on this lane's half)
#pragma unroll
    for (int r = 0; r < 8; ++r) {
      const float cand  = halfmax(fmaxf(fmaxf(s0[r], s1[r]), fmaxf(s2[r], s3[r])));
      const float mnew  = fmaxf(mrow[r], cand);
      const float alpha = __expf(mrow[r] - mnew);
      const float p0 = __expf(s0[r] - mnew);
      const float p1 = __expf(s1[r] - mnew);
      const float p2 = __expf(s2[r] - mnew);
      const float p3 = __expf(s3[r] - mnew);
      s0[r] = p0; s1[r] = p1; s2[r] = p2; s3[r] = p3;
      lrow[r] = lrow[r] * alpha + halfsum((p0 + p1) + (p2 + p3));
      mrow[r] = mnew;
      o0[r] *= alpha; o1[r] *= alpha; o2[r] *= alpha; o3[r] *= alpha;
    }

    // ---- transpose P: C layout -> A layout via per-wave LDS bounce
#pragma unroll
    for (int r = 0; r < 8; ++r) {
      const int rm = r + 8 * half;
      pbuf[wv][rm * 64 + row]      = (__bf16)s0[r];
      pbuf[wv][rm * 64 + 16 + row] = (__bf16)s1[r];
      pbuf[wv][rm * 64 + 32 + row] = (__bf16)s2[r];
      pbuf[wv][rm * 64 + 48 + row] = (__bf16)s3[r];
    }
    __syncthreads();
    const v16bf pa0 = ldv16_c(&pbuf[wv][row * 64 + half * 16]);        // keys 0..31
    const v16bf pa1 = ldv16_c(&pbuf[wv][row * 64 + 32 + half * 16]);   // keys 32..63
    __syncthreads();

    // ---- O += P * V  (V^T: slot i <-> key, contiguous 32B chunks)
#pragma unroll
    for (int c = 0; c < 4; ++c) {
      const __bf16* vb = Vt + (size_t)(c * 16 + row) * LL + kt * 64 + half * 16;
      const v16bf bv0 = ldv16_c(vb);
      const v16bf bv1 = ldv16_c(vb + 32);
      if (c == 0)      { o0 = wmma_bf16(pa0, bv0, o0); o0 = wmma_bf16(pa1, bv1, o0); }
      else if (c == 1) { o1 = wmma_bf16(pa0, bv0, o1); o1 = wmma_bf16(pa1, bv1, o1); }
      else if (c == 2) { o2 = wmma_bf16(pa0, bv0, o2); o2 = wmma_bf16(pa1, bv1, o2); }
      else             { o3 = wmma_bf16(pa0, bv0, o3); o3 = wmma_bf16(pa1, bv1, o3); }
    }
  }

  // ---- normalize + store [b, l, h*64+hd] bf16
#pragma unroll
  for (int r = 0; r < 8; ++r) {
    const float inv = 1.0f / lrow[r];
    o0[r] *= inv; o1[r] *= inv; o2[r] *= inv; o3[r] *= inv;
  }
#pragma unroll
  for (int r = 0; r < 8; ++r) {
    const int qrow = qt * 16 + r + 8 * half;
    const size_t base = (size_t)(bat * LL + qrow) * OC + hed * 64 + row;
    abf[base +  0] = (__bf16)o0[r];
    abf[base + 16] = (__bf16)o1[r];
    abf[base + 32] = (__bf16)o2[r];
    abf[base + 48] = (__bf16)o3[r];
  }
}

// ---------------------------------------------------------------------------
// Kernel 3: output projection  out = A(8192x512 bf16) * Wo + bo  (fp32 out)
// ---------------------------------------------------------------------------
__global__ __launch_bounds__(256)
void mha_oproj(const __bf16* __restrict__ A, const __bf16* __restrict__ po,
               const float* __restrict__ bo, float* __restrict__ out) {
  const int lane = threadIdx.x & 31;
  const int wv   = threadIdx.x >> 5;
  const int tile = blockIdx.x * 8 + wv;
  const int tn = tile & 31;
  const int tm = tile >> 5;
  const int row  = lane & 15;
  const int half = lane >> 4;
  const int col  = tn * 16 + row;
  const __bf16* Ar = A + (size_t)(tm * 16 + row) * OC;
  const __bf16* Pb = po + ((size_t)tn * 16 * 32 + lane) * 16;

  v8f acc = {};
#pragma unroll 4
  for (int k0 = 0; k0 < OC; k0 += 32) {
    const v16bf a = ldv16_a(Ar + k0 + half * 8);
    const v16bf b = ldv16_c(Pb + (size_t)(k0 >> 5) * 32 * 16);
    acc = wmma_bf16(a, b, acc);
  }

  const float bias = bo[col];
#pragma unroll
  for (int r = 0; r < 8; ++r) {
    const int n = tm * 16 + r + 8 * half;
    out[(size_t)n * DD + col] = acc[r] + bias;
  }
}

// ---------------------------------------------------------------------------
extern "C" void kernel_launch(void* const* d_in, const int* in_sizes, int n_in,
                              void* d_out, int out_size, void* d_ws,
                              size_t ws_size, hipStream_t stream) {
  const float* query = (const float*)d_in[0];
  const float* key_  = (const float*)d_in[1];
  const float* value = (const float*)d_in[2];
  const int*   mask  = (const int*)d_in[3];
  const float* Wq = (const float*)d_in[4];
  const float* bq = (const float*)d_in[5];
  const float* Wk = (const float*)d_in[6];
  const float* bk = (const float*)d_in[7];
  const float* Wv = (const float*)d_in[8];
  const float* bv = (const float*)d_in[9];
  const float* Wo = (const float*)d_in[10];
  const float* bo = (const float*)d_in[11];
  float* out = (float*)d_out;

  const size_t tsz = (size_t)BB * HH * LL * HDD;   // 4,194,304 elems / tensor
  const size_t psz = (size_t)32 * 16 * 32 * 16;    // 262,144 elems / packed W
  __bf16* qbf = (__bf16*)d_ws;
  __bf16* kbf = qbf + tsz;
  __bf16* vT  = kbf + tsz;       // transposed V [b,h,hd,l]
  __bf16* abf = vT + tsz;
  __bf16* pq  = abf + tsz;
  __bf16* pk  = pq + psz;
  __bf16* pv  = pk + psz;
  __bf16* po  = pv + psz;
  __bf16* xq  = po + psz;        // packed activations (A-operand chunks)
  __bf16* xk  = xq + tsz;
  __bf16* xv  = xk + tsz;        // total ~58 MB workspace

  pack_w<<<dim3(64, 4), dim3(256), 0, stream>>>(Wq, Wk, Wv, Wo, pq, pk, pv, po);
  pack_x<<<dim3(1024, 3), dim3(256), 0, stream>>>(query, key_, value, xq, xk, xv);

  mha_proj_qkv<<<dim3(2048, 3), dim3(256), 0, stream>>>(
      xq, xk, xv, pq, bq, pk, bk, pv, bv, qbf, kbf, vT);

  mha_flash<<<dim3(512), dim3(256), 0, stream>>>(qbf, kbf, vT, mask, abf);

  mha_oproj<<<dim3(2048), dim3(256), 0, stream>>>(abf, po, bo, out);
}
